// LlamaSdpaAttention_84482006712585
// MI455X (gfx1250) — matrix operand order repaired and checked by reading, not compiled, not run on hardware
//
#include <hip/hip_runtime.h>
#include <stddef.h>

// ---------------------------------------------------------------------------
// MI455X (gfx1250) Llama SDPA attention block.
// All matmul work on v_wmma_f32_16x16x32_bf16 (wave32 WMMA), fp32 accum.
// GEMMs: LDS double-buffered with async global->LDS copies (ASYNCcnt path).
// Attention: flash-style streaming (non-causal; mask is all-true).
// ---------------------------------------------------------------------------

typedef unsigned short u16_t;
typedef __bf16 v8bf  __attribute__((ext_vector_type(8)));
typedef __bf16 v16bf __attribute__((ext_vector_type(16)));
typedef float  v8f   __attribute__((ext_vector_type(8)));
typedef int    v4i_t __attribute__((ext_vector_type(4)));

typedef __attribute__((address_space(1))) v4i_t* g_v4i_p;  // global int4*
typedef __attribute__((address_space(3))) v4i_t* l_v4i_p;  // LDS int4*

#define B_  2
#define S_  2048
#define D_  2048
#define H_  16
#define HD_ 128

#if __has_builtin(__builtin_amdgcn_global_load_async_to_lds_b128)
#define ASYNC_LDS 1
#else
#define ASYNC_LDS 0
#endif

// fp32 -> bf16, round-to-nearest-even (bit-level; no __bf16 arithmetic needed)
__device__ __forceinline__ u16_t f2bf(float f) {
  union { float f; unsigned u; } v; v.f = f;
  unsigned u = v.u;
  return (u16_t)((u + 0x7fffu + ((u >> 16) & 1u)) >> 16);
}

union FragU { v16bf v; v8bf h[2]; };

// 16B global -> LDS copy; async (tracked by ASYNCcnt) when available.
__device__ __forceinline__ void cp16(u16_t* l, const u16_t* g) {
#if ASYNC_LDS
  __builtin_amdgcn_global_load_async_to_lds_b128(
      (g_v4i_p)const_cast<u16_t*>(g), (l_v4i_p)l, 0, 0);
#else
  *(uint4*)l = *(const uint4*)g;
#endif
}

__device__ __forceinline__ void async_wait0() {
#if ASYNC_LDS
#if __has_builtin(__builtin_amdgcn_s_wait_asynccnt)
  __builtin_amdgcn_s_wait_asynccnt(0);
#else
  asm volatile("s_wait_asynccnt 0" ::: "memory");
#endif
#endif
}

// Vectorized fp32 -> bf16 cast: float4 in, 2 packed dwords out.
__global__ void cast_f32_to_bf16_v4(const float4* __restrict__ in,
                                    uint2* __restrict__ out, int n4) {
  for (int i = blockIdx.x * blockDim.x + threadIdx.x; i < n4;
       i += gridDim.x * blockDim.x) {
    float4 f = in[i];
    uint2 o;
    o.x = (unsigned)f2bf(f.x) | ((unsigned)f2bf(f.y) << 16);
    o.y = (unsigned)f2bf(f.z) | ((unsigned)f2bf(f.w) << 16);
    out[i] = o;
  }
}

// ---------------------------------------------------------------------------
// Tiled NT GEMM: C[m,n] = sum_k A[m,k] * W[n,k]   (M=4096, N=2048, K=2048)
// Block: 256 threads (8 waves). Workgroup tile 64(M) x 128(N), BK=32,
// double-buffered LDS with async staging. Wave tile 32x32 = 2x2 WMMA tiles.
// Epilogues:
//   EPI_ROPE : rotary embed, bf16 store to [B,H,S,hd]
//   EPI_VT   : bf16 store transposed to [B,H,hd,S]
//   EPI_F32  : fp32 store row-major [M,N]  (final output)
// ---------------------------------------------------------------------------
enum { EPI_ROPE = 0, EPI_VT = 1, EPI_F32 = 2 };

template <int EPI>
__global__ __launch_bounds__(256)
void gemm_nt_bf16(const u16_t* __restrict__ A, const u16_t* __restrict__ Bw,
                  void* __restrict__ outv) {
  __shared__ u16_t As[2][64 * 32];    // [buf][m][k]
  __shared__ u16_t Bs[2][128 * 32];   // [buf][n][k]

  const int tid  = threadIdx.x;
  const int lane = tid & 31;
  const int wave = tid >> 5;
  const int wm   = wave >> 2;      // 0..1 : 32-row slab
  const int wn   = wave & 3;       // 0..3 : 32-col slab
  const int l15  = lane & 15;
  const int hlf  = lane >> 4;      // lane half (0/1)
  const int bn   = blockIdx.x * 128;
  const int bm   = blockIdx.y * 64;

  // staging assignments (16B per copy)
  const int ar  = tid >> 2;               // 0..63
  const int ac8 = (tid & 3) * 8;
  const size_t a_row = (size_t)(bm + ar) * D_;
  const int br0 = (2 * tid) >> 2,     bc0 = ((2 * tid) & 3) * 8;
  const int br1 = (2 * tid + 1) >> 2, bc1 = ((2 * tid + 1) & 3) * 8;
  const size_t b_row0 = (size_t)(bn + br0) * D_;
  const size_t b_row1 = (size_t)(bn + br1) * D_;

  v8f acc[2][2] = {};

  // prologue: stage k-step 0 into buffer 0
  cp16(As[0] + ar * 32 + ac8, A + a_row + 0 + ac8);
  cp16(Bs[0] + br0 * 32 + bc0, Bw + b_row0 + 0 + bc0);
  cp16(Bs[0] + br1 * 32 + bc1, Bw + b_row1 + 0 + bc1);
  async_wait0();
  __syncthreads();

  int buf = 0;
  for (int k0 = 0; k0 < D_; k0 += 32) {
    // stage next K-step into the other buffer while we compute
    if (k0 + 32 < D_) {
      const int kn = k0 + 32;
      cp16(As[buf ^ 1] + ar * 32 + ac8, A + a_row + kn + ac8);
      cp16(Bs[buf ^ 1] + br0 * 32 + bc0, Bw + b_row0 + kn + bc0);
      cp16(Bs[buf ^ 1] + br1 * 32 + bc1, Bw + b_row1 + kn + bc1);
    }

    // A fragment (16x32 bf16): lane l15 = row, two 8-half chunks:
    //   k = hlf*16 + [0..7]  and  k = hlf*16 + 8 + [0..7]
    FragU af[2]; v16bf bfr[2];
#pragma unroll
    for (int ti = 0; ti < 2; ++ti) {
      const u16_t* ap = As[buf] + (wm * 32 + ti * 16 + l15) * 32;
      af[ti].h[0] = *(const v8bf*)(ap + hlf * 16);
      af[ti].h[1] = *(const v8bf*)(ap + hlf * 16 + 8);
    }
    // B fragment (32x16 bf16): lane l15 = n, k = hlf*16 + [0..15] contiguous
#pragma unroll
    for (int tj = 0; tj < 2; ++tj)
      bfr[tj] = *(const v16bf*)(Bs[buf] + (wn * 32 + tj * 16 + l15) * 32 + hlf * 16);

#pragma unroll
    for (int ti = 0; ti < 2; ++ti)
#pragma unroll
      for (int tj = 0; tj < 2; ++tj)
        acc[ti][tj] = __builtin_amdgcn_wmma_f32_16x16x32_bf16(
            false, af[ti].v, false, bfr[tj], (short)0, acc[ti][tj],
            false, false);

    async_wait0();
    __syncthreads();
    buf ^= 1;
  }

  // Epilogue. C layout: VGPR r -> m = r + 8*hlf ; n = l15 (per 16x16 tile).
#pragma unroll
  for (int ti = 0; ti < 2; ++ti) {
#pragma unroll
    for (int tj = 0; tj < 2; ++tj) {
#pragma unroll
      for (int r = 0; r < 8; ++r) {
        const int m = bm + wm * 32 + ti * 16 + r + hlf * 8;
        const int n = bn + wn * 32 + tj * 16 + l15;
        float x = acc[ti][tj][r];
        if (EPI == EPI_ROPE) {
          // interleaved rotation: partner element lives in lane^1 (n^1)
          float p = __shfl_xor(x, 1, 32);
          const int s  = m & (S_ - 1);
          const int d  = n & (HD_ - 1);
          const int fi = d & 63;
          // inv_freq = 10000^(-fi/64) = exp(-fi * ln(10000)/64)
          float ang = (float)s * __expf((float)fi * -0.14391157f);
          float sn, cs;
          __sincosf(ang, &sn, &cs);
          float o = (n & 1) ? (x * cs + p * sn) : (x * cs - p * sn);
          const int b = m >> 11, h = n >> 7;
          ((u16_t*)outv)[(((size_t)(b * H_ + h) * S_ + s) * HD_) + d] = f2bf(o);
        } else if (EPI == EPI_VT) {
          const int s = m & (S_ - 1), d = n & (HD_ - 1);
          const int b = m >> 11, h = n >> 7;
          ((u16_t*)outv)[(((size_t)(b * H_ + h) * HD_ + d) * S_) + s] = f2bf(x);
        } else {
          ((float*)outv)[(size_t)m * D_ + n] = x;
        }
      }
    }
  }
}

// ---------------------------------------------------------------------------
// Flash attention (non-causal, mask all-true).
// Grid: B*H*(S/128) blocks, 256 threads (8 waves). Each wave: 16 q-rows.
// Per iteration: 32 k-positions -> 8 QK^T WMMA (d=128 contraction),
// online softmax (fp32), P via LDS relayout, 8 PV WMMA into 16x128 fp32 acc.
//   Qr, Kr : bf16 [B,H,S,hd]   Vt : bf16 [B,H,hd,S]   Ob : bf16 [B,S,D]
// ---------------------------------------------------------------------------
__global__ __launch_bounds__(256)
void attn_flash(const u16_t* __restrict__ Qr, const u16_t* __restrict__ Kr,
                const u16_t* __restrict__ Vt, u16_t* __restrict__ Ob) {
  __shared__ u16_t pbuf[8 * 16 * 32];   // per-wave P tile [16 q][32 k]

  const int lane = threadIdx.x & 31;
  const int wave = threadIdx.x >> 5;
  const int l15  = lane & 15;
  const int hlf  = lane >> 4;

  const int qchunks = S_ / 128;                 // 16
  const int qc = blockIdx.x % qchunks;
  const int bh = blockIdx.x / qchunks;          // b*H + h
  const int b  = bh >> 4, h = bh & 15;
  const int q0 = qc * 128 + wave * 16;

  const u16_t* Qp = Qr + ((size_t)bh * S_ + q0) * HD_;
  const u16_t* Kp = Kr + (size_t)bh * S_ * HD_;
  const u16_t* Vp = Vt + (size_t)bh * HD_ * S_;
  u16_t* pb = pbuf + wave * 16 * 32;

  // Preload Q fragments: 4 K-steps of the d=128 contraction
  FragU qf[4];
#pragma unroll
  for (int st = 0; st < 4; ++st) {
    const u16_t* qrow = Qp + (size_t)l15 * HD_ + st * 32;
    qf[st].h[0] = *(const v8bf*)(qrow + hlf * 16);
    qf[st].h[1] = *(const v8bf*)(qrow + hlf * 16 + 8);
  }

  v8f acc[8] = {};            // 16 q-rows x 128 d, fp32
  float mrow[8], lrow[8];
#pragma unroll
  for (int r = 0; r < 8; ++r) { mrow[r] = -1e30f; lrow[r] = 0.f; }

  const float scale = 0.08838834764831845f;     // 1/sqrt(128)

  for (int kt = 0; kt < S_; kt += 32) {
    // L2 prefetch hint for the next K/V tiles (global_prefetch_b8)
    if (kt + 32 < S_) {
      __builtin_prefetch(Kp + (size_t)(kt + 32 + l15) * HD_, 0, 1);
      __builtin_prefetch(Vp + (size_t)l15 * S_ + kt + 32, 0, 1);
    }

    // --- scores: two 16q x 16k tiles over kpos [kt, kt+32) ---
    v8f s0 = {}, s1 = {};
#pragma unroll
    for (int st = 0; st < 4; ++st) {
      // B frag: lane = kpos, 16 contiguous d values from the K row
      v16bf k0f = *(const v16bf*)(Kp + (size_t)(kt + l15) * HD_ + st * 32 + hlf * 16);
      v16bf k1f = *(const v16bf*)(Kp + (size_t)(kt + 16 + l15) * HD_ + st * 32 + hlf * 16);
      s0 = __builtin_amdgcn_wmma_f32_16x16x32_bf16(false, qf[st].v, false, k0f,
                                                   (short)0, s0, false, false);
      s1 = __builtin_amdgcn_wmma_f32_16x16x32_bf16(false, qf[st].v, false, k1f,
                                                   (short)0, s1, false, false);
    }

    // --- online softmax stats (rows live across 16-lane halves) ---
    float p0[8], p1[8], al[8];
#pragma unroll
    for (int r = 0; r < 8; ++r) {
      float a  = s0[r] * scale;
      float c  = s1[r] * scale;
      float mx = fmaxf(a, c);
      mx = fmaxf(mx, __shfl_xor(mx, 1, 32));
      mx = fmaxf(mx, __shfl_xor(mx, 2, 32));
      mx = fmaxf(mx, __shfl_xor(mx, 4, 32));
      mx = fmaxf(mx, __shfl_xor(mx, 8, 32));
      float mn = fmaxf(mrow[r], mx);
      float alpha = __expf(mrow[r] - mn);
      mrow[r] = mn;
      p0[r] = __expf(a - mn);
      p1[r] = __expf(c - mn);
      float rs = p0[r] + p1[r];
      rs += __shfl_xor(rs, 1, 32);
      rs += __shfl_xor(rs, 2, 32);
      rs += __shfl_xor(rs, 4, 32);
      rs += __shfl_xor(rs, 8, 32);
      lrow[r] = lrow[r] * alpha + rs;
      al[r] = alpha;
    }

    // --- relayout P (C-layout -> A-layout) through LDS ---
#pragma unroll
    for (int r = 0; r < 8; ++r) {
      const int row = r + hlf * 8;
      pb[row * 32 + l15]      = f2bf(p0[r]);
      pb[row * 32 + 16 + l15] = f2bf(p1[r]);
    }
    __syncthreads();
    FragU pf;
    pf.h[0] = *(const v8bf*)(pb + l15 * 32 + hlf * 16);
    pf.h[1] = *(const v8bf*)(pb + l15 * 32 + hlf * 16 + 8);

    // --- acc = acc*alpha + P @ V  (Vt rows are contiguous in kpos) ---
#pragma unroll
    for (int nt = 0; nt < 8; ++nt) {
#pragma unroll
      for (int r = 0; r < 8; ++r) acc[nt][r] *= al[r];
      v16bf vf = *(const v16bf*)(Vp + (size_t)(nt * 16 + l15) * S_ + kt + hlf * 16);
      acc[nt] = __builtin_amdgcn_wmma_f32_16x16x32_bf16(false, pf.v, false, vf,
                                                        (short)0, acc[nt],
                                                        false, false);
    }
    __syncthreads();
  }

  // Normalize and store bf16 to Ob[b, s, h*128 + d] (row-major [B*S, D])
#pragma unroll
  for (int nt = 0; nt < 8; ++nt) {
#pragma unroll
    for (int r = 0; r < 8; ++r) {
      const int row = r + hlf * 8;
      const size_t off = (size_t)(b * S_ + q0 + row) * D_ + h * HD_ + nt * 16 + l15;
      Ob[off] = f2bf(acc[nt][r] / lrow[r]);
    }
  }
}

// ---------------------------------------------------------------------------
extern "C" void kernel_launch(void* const* d_in, const int* in_sizes, int n_in,
                              void* d_out, int out_size, void* d_ws, size_t ws_size,
                              hipStream_t stream) {
  (void)in_sizes; (void)n_in; (void)out_size; (void)ws_size;
  const float* hs = (const float*)d_in[0];
  const float* Wq = (const float*)d_in[1];
  const float* Wk = (const float*)d_in[2];
  const float* Wv = (const float*)d_in[3];
  const float* Wo = (const float*)d_in[4];
  // d_in[5] = attention_mask (all true), d_in[6] = position_ids (== S): unused.

  u16_t* w = (u16_t*)d_ws;
  const size_t HS = (size_t)B_ * S_ * D_;   // 8,388,608 elems
  const size_t WW = (size_t)D_ * D_;        // 4,194,304 elems
  u16_t* hsb = w;            // bf16 hidden states [B*S, D]
  u16_t* wqb = hsb + HS;
  u16_t* wkb = wqb + WW;
  u16_t* wvb = wkb + WW;
  u16_t* wob = wvb + WW;
  u16_t* Qr  = wob + WW;     // [B,H,S,hd] (RoPE'd)
  u16_t* Kr  = Qr + HS;      // [B,H,S,hd] (RoPE'd)
  u16_t* Vt  = Kr + HS;      // [B,H,hd,S]
  u16_t* Ob  = hsb;          // attention out [B*S, D]; reuses dead hsb

  dim3 blk(256);
  cast_f32_to_bf16_v4<<<2048, blk, 0, stream>>>((const float4*)hs, (uint2*)hsb,
                                                (int)(HS / 4));
  cast_f32_to_bf16_v4<<<1024, blk, 0, stream>>>((const float4*)Wq, (uint2*)wqb,
                                                (int)(WW / 4));
  cast_f32_to_bf16_v4<<<1024, blk, 0, stream>>>((const float4*)Wk, (uint2*)wkb,
                                                (int)(WW / 4));
  cast_f32_to_bf16_v4<<<1024, blk, 0, stream>>>((const float4*)Wv, (uint2*)wvb,
                                                (int)(WW / 4));
  cast_f32_to_bf16_v4<<<1024, blk, 0, stream>>>((const float4*)Wo, (uint2*)wob,
                                                (int)(WW / 4));

  dim3 gg(D_ / 128, (B_ * S_) / 64);   // (16, 64)
  gemm_nt_bf16<EPI_ROPE><<<gg, blk, 0, stream>>>(hsb, wqb, (void*)Qr);
  gemm_nt_bf16<EPI_ROPE><<<gg, blk, 0, stream>>>(hsb, wkb, (void*)Kr);
  gemm_nt_bf16<EPI_VT  ><<<gg, blk, 0, stream>>>(hsb, wvb, (void*)Vt);

  attn_flash<<<B_ * H_ * (S_ / 128), blk, 0, stream>>>(Qr, Kr, Vt, Ob);

  gemm_nt_bf16<EPI_F32><<<gg, blk, 0, stream>>>(Ob, wob, d_out);
}
